// CalculateFlow_62801011802527
// MI455X (gfx1250) — compile-verified, matrix operand order checked by reading, not run on hardware
//
#include <hip/hip_runtime.h>
#include <hip/hip_bf16.h>

// Problem constants (match setup_inputs: 544x960, T=8, R=4)
constexpr int IH = 544;
constexpr int IW = 960;
constexpr int TT = 8;       // template size
constexpr int RR = 4;       // search range
constexpr int HB = IH / TT; // 68
constexpr int WB = IW / TT; // 120
constexpr int NB = HB * WB; // 8160 blocks
constexpr int PW = IW + 2 * RR; // 968 padded prev width
constexpr int PH = IH + 2 * RR; // 552
constexpr int NC = (2 * RR + 1) * (2 * RR + 1); // 81 candidates

typedef __attribute__((ext_vector_type(8))) int v8i;

// ---------------------------------------------------------------------------
// Binomial blur ([1 2 1]/4 separable, replicate pad), evaluated at (y,x)
// ---------------------------------------------------------------------------
__device__ __forceinline__ float vblur(const float* __restrict__ img, int y, int x) {
    int ym = y > 0 ? y - 1 : 0;
    int yp = y < IH - 1 ? y + 1 : IH - 1;
    return (img[(size_t)ym * IW + x] + 2.0f * img[(size_t)y * IW + x] +
            img[(size_t)yp * IW + x]) * 0.25f;
}
__device__ __forceinline__ float blur_at(const float* __restrict__ img, int y, int x) {
    int xm = x > 0 ? x - 1 : 0;
    int xp = x < IW - 1 ? x + 1 : IW - 1;
    return (vblur(img, y, xm) + 2.0f * vblur(img, y, x) + vblur(img, y, xp)) * 0.25f;
}
__device__ __forceinline__ unsigned char quant8(float b) {
    float q = rintf(b * 255.0f);
    q = fminf(fmaxf(q, 0.0f), 255.0f);
    return (unsigned char)(int)q;
}

// ---------------------------------------------------------------------------
// Kernel 1: blur+quantize current image; also keep float blur for gradients
// ---------------------------------------------------------------------------
__global__ __launch_bounds__(256) void prep_cur_kernel(const float* __restrict__ img,
                                                       float* __restrict__ cur_blur,
                                                       unsigned char* __restrict__ cur_q) {
    int idx = blockIdx.x * blockDim.x + threadIdx.x;
    if (idx >= IH * IW) return;
    int y = idx / IW, x = idx % IW;
    float b = blur_at(img, y, x);
    cur_blur[idx] = b;
    cur_q[idx] = quant8(b);
}

// ---------------------------------------------------------------------------
// Kernel 2: blur+quantize prev image into replicate-padded (PH x PW) u8
// ---------------------------------------------------------------------------
__global__ __launch_bounds__(256) void prep_prev_kernel(const float* __restrict__ img,
                                                        unsigned char* __restrict__ prev_pad) {
    int idx = blockIdx.x * blockDim.x + threadIdx.x;
    if (idx >= PH * PW) return;
    int py = idx / PW, px = idx % PW;
    int y = py - RR; y = y < 0 ? 0 : (y > IH - 1 ? IH - 1 : y);
    int x = px - RR; x = x < 0 ? 0 : (x > IW - 1 ? IW - 1 : x);
    prev_pad[idx] = quant8(blur_at(img, y, x));
}

// ---------------------------------------------------------------------------
// Kernel 3: SAD cost volume via V_WMMA_I32_16X16X64_IU8 + spiral argmin.
// One wave per 8x8 block; 16 candidates per WMMA (A = packed |cur-prev| u8,
// B = all-ones u8 -> D[m][*] = SAD of candidate m). 6 WMMAs cover 81 cands.
// Search window staged into LDS with GLOBAL_LOAD_ASYNC_TO_LDS (ASYNCcnt).
// ---------------------------------------------------------------------------
__global__ __launch_bounds__(256) void sad_kernel(const unsigned char* __restrict__ cur_q,
                                                  const unsigned char* __restrict__ prev_pad,
                                                  float* __restrict__ vec) {
    __shared__ __align__(16) unsigned char win[8][16 * 16]; // per-wave 16x16 window
    __shared__ int spir[NC];

    const int tid = threadIdx.x;
    const int lane = tid & 31;
    const int wv = tid >> 5;
    const int blk = blockIdx.x * 8 + wv;          // grid = NB/8 exactly
    const int by = blk / WB, bx = blk % WB;

    // Async-stage the 16x16 u8 search window: each lane moves 8 bytes
    // (lane -> row = lane/2, half = lane&1). One async op, tracked by ASYNCcnt.
    {
        const int row = lane >> 1;
        const int half = lane & 1;
        const size_t wrow0 = (size_t)(by * TT) * PW + (size_t)(bx * TT);
        const unsigned char* gsrc = prev_pad + wrow0 + (size_t)row * PW + (size_t)(half * 8);
        unsigned ldsoff = (unsigned)(size_t)(&win[wv][row * 16 + half * 8]);
        asm volatile("global_load_async_to_lds_b64 %0, %1, off"
                     :
                     : "v"(ldsoff), "v"(gsrc)
                     : "memory");
    }

    // Fill spiral tie-break table (replicates reference spiral_pattern)
    if (tid == 0) {
        int n = 0;
        for (int dy = 0; dy <= RR; ++dy)
            for (int dir = 0; dir < 4; ++dir) {
                int limit = (dir == 0 && dy == 0) ? 1 : dy;
                for (int dx = -dy; dx < limit; ++dx) {
                    int wx, wy;
                    if (dir == 0)      { wx = RR + dx; wy = RR + dy; }
                    else if (dir == 1) { wx = RR + dy; wy = RR - dx; }
                    else if (dir == 2) { wx = RR - dx; wy = RR - dy; }
                    else               { wx = RR - dy; wy = RR + dx; }
                    ++n;
                    spir[wy * (2 * RR + 1) + wx] = n;
                }
            }
    }
    __syncthreads();

    // Template rows owned by this lane per the 8-bit A-matrix layout:
    // lane L (<16) holds rows ty = {0,2,4,6}; lane L+16 holds {1,3,5,7}.
    const int hi = lane >> 4;        // 0 or 1
    unsigned tmpl[8];
    for (int r = 0; r < 4; ++r) {
        int y = by * TT + 2 * r + hi;
        uint2 tr = *(const uint2*)(cur_q + (size_t)y * IW + (size_t)(bx * TT));
        tmpl[2 * r] = tr.x;
        tmpl[2 * r + 1] = tr.y;
    }

    // Window must be resident in LDS before the first ds read.
    asm volatile("s_wait_asynccnt 0" ::: "memory");

    const int m = lane & 15;
    const v8i bones = {0x01010101, 0x01010101, 0x01010101, 0x01010101,
                       0x01010101, 0x01010101, 0x01010101, 0x01010101};
    int bestKey = 0x7FFFFFFF;

    for (int g = 0; g < 6; ++g) {
        int c = g * 16 + m;
        int ce = c > (NC - 1) ? (NC - 1) : c;      // clamp pad lanes in last group
        int dy = ce / 9 - RR;
        int dx = ce % 9 - RR;

        // Build A row (candidate m): packed u8 abs-diffs, K split per layout
        v8i A;
        const unsigned char* wp = &win[wv][0];
        for (int r = 0; r < 4; ++r) {
            int ty = 2 * r + hi;
            const unsigned char* pr = wp + (ty + dy + RR) * 16 + (dx + RR);
            unsigned lo = 0, hw = 0;
            for (int j = 0; j < 4; ++j) {
                int tb = (int)((tmpl[2 * r] >> (8 * j)) & 0xFF);
                int pb = (int)pr[j];
                int ad = tb - pb; ad = ad < 0 ? -ad : ad;
                lo |= (unsigned)ad << (8 * j);
            }
            for (int j = 0; j < 4; ++j) {
                int tb = (int)((tmpl[2 * r + 1] >> (8 * j)) & 0xFF);
                int pb = (int)pr[4 + j];
                int ad = tb - pb; ad = ad < 0 ? -ad : ad;
                hw |= (unsigned)ad << (8 * j);
            }
            A[2 * r] = (int)lo;
            A[2 * r + 1] = (int)hw;
        }

        v8i Cz = {0, 0, 0, 0, 0, 0, 0, 0};
        // D[m][n] = sum_k |a-b| ; unsigned A/B
        v8i D = __builtin_amdgcn_wmma_i32_16x16x64_iu8(false, A, false, bones,
                                                       Cz, false, false);

        // D layout: VGPR j, lanes 0-15 -> M=j ; lanes 16-31 -> M=j+8.
        // All N columns identical (B = ones). Uniform scalar reduction:
        for (int j = 0; j < 8; ++j) {
            int sLo = __builtin_amdgcn_readlane(D[j], 0);   // cand g*16 + j
            int sHi = __builtin_amdgcn_readlane(D[j], 16);  // cand g*16 + 8 + j
            int c0 = g * 16 + j;
            if (c0 < NC) {
                int key = (sLo << 14) | (spir[c0] << 7) | c0;
                bestKey = key < bestKey ? key : bestKey;
            }
            int c1 = c0 + 8;
            if (c1 < NC) {
                int key = (sHi << 14) | (spir[c1] << 7) | c1;
                bestKey = key < bestKey ? key : bestKey;
            }
        }
    }

    int bc = bestKey & 127;
    int bdy = bc / 9 - RR;
    int bdx = bc % 9 - RR;
    if (lane == 0) {
        vec[blk] = (float)(-bdy);        // y component (negated offset)
        vec[NB + blk] = (float)(-bdx);   // x component
    }
}

// ---------------------------------------------------------------------------
// Kernel 4: 3x3 median filter (replicate pad) per component
// ---------------------------------------------------------------------------
__device__ __forceinline__ void mnmx(float& a, float& b) {
    float t = fminf(a, b);
    b = fmaxf(a, b);
    a = t;
}
__device__ __forceinline__ float median9(float p0, float p1, float p2,
                                         float p3, float p4, float p5,
                                         float p6, float p7, float p8) {
    mnmx(p1, p2); mnmx(p4, p5); mnmx(p7, p8);
    mnmx(p0, p1); mnmx(p3, p4); mnmx(p6, p7);
    mnmx(p1, p2); mnmx(p4, p5); mnmx(p7, p8);
    mnmx(p0, p3); mnmx(p5, p8); mnmx(p4, p7);
    mnmx(p3, p6); mnmx(p1, p4); mnmx(p2, p5);
    mnmx(p4, p7); mnmx(p4, p2); mnmx(p6, p4);
    mnmx(p4, p2);
    return p4;
}
__global__ __launch_bounds__(256) void median_kernel(const float* __restrict__ vec,
                                                     float* __restrict__ vmed) {
    int blk = blockIdx.x * blockDim.x + threadIdx.x;
    if (blk >= NB) return;
    int by = blk / WB, bx = blk % WB;
    for (int comp = 0; comp < 2; ++comp) {
        const float* v = vec + comp * NB;
        float p[9];
        int n = 0;
        for (int i = -1; i <= 1; ++i)
            for (int j = -1; j <= 1; ++j) {
                int yy = by + i; yy = yy < 0 ? 0 : (yy > HB - 1 ? HB - 1 : yy);
                int xx = bx + j; xx = xx < 0 ? 0 : (xx > WB - 1 ? WB - 1 : xx);
                p[n++] = v[yy * WB + xx];
            }
        vmed[comp * NB + blk] = median9(p[0], p[1], p[2], p[3], p[4],
                                        p[5], p[6], p[7], p[8]);
    }
}

// ---------------------------------------------------------------------------
// Kernel 5: Lucas-Kanade subpixel refinement on the block border ring
// ---------------------------------------------------------------------------
__global__ __launch_bounds__(256) void lk_kernel(const float* __restrict__ cur_blur,
                                                 const unsigned char* __restrict__ cur_q,
                                                 const unsigned char* __restrict__ prev_pad,
                                                 const float* __restrict__ vmed,
                                                 float* __restrict__ out) {
    int blk = blockIdx.x * blockDim.x + threadIdx.x;
    if (blk >= NB) return;
    int by = blk / WB, bx = blk % WB;

    float vy = vmed[blk];
    float vx = vmed[NB + blk];

    float ry = rintf(vy); ry = fminf(fmaxf(ry, (float)-RR), (float)RR);
    float rx = rintf(vx); rx = fminf(fmaxf(rx, (float)-RR), (float)RR);
    int off_y = -(int)ry;        // vec stores negated offsets
    int off_x = -(int)rx;

    const float inv255 = 1.0f / 255.0f;
    float a = 0.f, b = 0.f, d = 0.f, pp = 0.f, qq = 0.f;
    for (int ty = 0; ty < TT; ++ty) {
        for (int tx = 0; tx < TT; ++tx) {
            if (ty > 0 && ty < TT - 1 && tx > 0 && tx < TT - 1) continue; // pred=0
            int y = by * TT + ty, x = bx * TT + tx;
            int xm = x > 0 ? x - 1 : 0, xp = x < IW - 1 ? x + 1 : IW - 1;
            int ym = y > 0 ? y - 1 : 0, yp = y < IH - 1 ? y + 1 : IH - 1;
            float gxv = (cur_blur[(size_t)y * IW + xp] - cur_blur[(size_t)y * IW + xm]) * 0.5f;
            float gyv = (cur_blur[(size_t)yp * IW + x] - cur_blur[(size_t)ym * IW + x]) * 0.5f;
            float tm = (float)cur_q[(size_t)y * IW + x] * inv255;
            float mt = (float)prev_pad[(size_t)(y + off_y + RR) * PW + (x + off_x + RR)] * inv255;
            float df = mt - tm;
            a += gxv * gxv;
            b += gxv * gyv;
            d += gyv * gyv;
            pp += df * gxv;
            qq += df * gyv;
        }
    }
    float det = a * d - b * b;
    bool bad = det <= 1e-6f;
    float safe = bad ? 1.0f : det;
    float su = (d * pp - b * qq) / safe;   // x component
    float sv = (a * qq - b * pp) / safe;   // y component
    float svo = (bad || fabsf(sv) >= 1.0f) ? 0.0f : sv;
    float suo = (bad || fabsf(su) >= 1.0f) ? 0.0f : su;

    out[blk] = vy + svo;
    out[NB + blk] = vx + suo;
}

// ---------------------------------------------------------------------------
// Host launcher
// ---------------------------------------------------------------------------
extern "C" void kernel_launch(void* const* d_in, const int* in_sizes, int n_in,
                              void* d_out, int out_size, void* d_ws, size_t ws_size,
                              hipStream_t stream) {
    (void)in_sizes; (void)n_in; (void)out_size; (void)ws_size;
    const float* img_cur = (const float*)d_in[0];
    const float* img_prev = (const float*)d_in[1];

    char* ws = (char*)d_ws;
    // workspace layout (all offsets 64B aligned)
    float* cur_blur        = (float*)(ws + 0);                       // 2,088,960 B
    unsigned char* cur_q   = (unsigned char*)(ws + 2088960);         //   522,240 B
    unsigned char* prev_pd = (unsigned char*)(ws + 2611200);         //   534,336 B
    float* vec             = (float*)(ws + 3145536);                 //    65,280 B
    float* vmed            = (float*)(ws + 3210816);                 //    65,280 B
    float* out             = (float*)d_out;

    prep_cur_kernel<<<(IH * IW + 255) / 256, 256, 0, stream>>>(img_cur, cur_blur, cur_q);
    prep_prev_kernel<<<(PH * PW + 255) / 256, 256, 0, stream>>>(img_prev, prev_pd);
    sad_kernel<<<NB / 8, 256, 0, stream>>>(cur_q, prev_pd, vec);
    median_kernel<<<(NB + 255) / 256, 256, 0, stream>>>(vec, vmed);
    lk_kernel<<<(NB + 255) / 256, 256, 0, stream>>>(cur_blur, cur_q, prev_pd, vmed, out);
}
